// ModelTorch_87333864997453
// MI455X (gfx1250) — compile-verified, matrix operand order checked by reading, not run on hardware
//
#include <hip/hip_runtime.h>

typedef __attribute__((ext_vector_type(8))) int   v8i;
typedef __attribute__((ext_vector_type(4))) int   v4i;
typedef __attribute__((ext_vector_type(4))) float v4f;

// One block (256 threads = 8 wave32) per row of the [B, D=8192] mask.
//
// Phase 1: stream the row in with non-temporal b128 loads (zero reuse, the
//          1 GB working set is 5x the 192 MB L2 -> NT avoids thrash), packing
//          the 0/1 mask values into bytes: 8 dwords/thread = WMMA A operand.
// Phase 2: per-wave popcount of 1024 mask bytes with ONE
//          v_wmma_i32_16x16x64_iu8 (B = all-one bytes => D[m][n] = rowsum_m).
//          After summing the 8 D VGPRs per lane, lane 0 holds rowsums 0..7 and
//          lane 16 holds rowsums 8..15, so readlane(0)+readlane(16) is the
//          wave's exact popcount. One LDS round + one barrier combines waves.
// Phase 3: every thread computes the page-aligned clear window (uniform) and
//          streams the floats back out from the packed registers (NT stores).
__global__ __launch_bounds__(256) void evict_mask_page_clear(
    const int* __restrict__ seq_lens,
    const int* __restrict__ evict_mask,
    const int* __restrict__ page_size_ptr,
    float* __restrict__ out,
    int D)
{
    const int row = blockIdx.x;
    const int tid = threadIdx.x;
    const size_t base = (size_t)row * (size_t)D;
    const v4i* __restrict__ mrow = (const v4i*)(evict_mask + base);
    v4f* __restrict__ orow = (v4f*)(out + base);

    // D == 8192: 8 chunks x 4 elements per thread, byte-packed.
    unsigned pk[8];
    v8i a;
#pragma unroll
    for (int c = 0; c < 8; ++c) {
        v4i m = __builtin_nontemporal_load(&mrow[c * 256 + tid]);
        unsigned p = (m.x != 0 ? 0x00000001u : 0u)
                   | (m.y != 0 ? 0x00000100u : 0u)
                   | (m.z != 0 ? 0x00010000u : 0u)
                   | (m.w != 0 ? 0x01000000u : 0u);
        pk[c] = p;
        a[c] = (int)p;
    }

    // Wave-level popcount of 1024 bytes in one matrix op.
    v8i bones;
#pragma unroll
    for (int c = 0; c < 8; ++c) bones[c] = 0x01010101;
    v8i cz = {};
    // (sgn_a, A, sgn_b, B, C, reuse_a, reuse_b)
    v8i dd = __builtin_amdgcn_wmma_i32_16x16x64_iu8(false, a, false, bones, cz,
                                                    false, false);
    int lanesum = 0;
#pragma unroll
    for (int c = 0; c < 8; ++c) lanesum += dd[c];

    // lane 0: sum of rowsums M=0..7; lane 16: sum of rowsums M=8..15.
    const int wavesum = __builtin_amdgcn_readlane(lanesum, 0) +
                        __builtin_amdgcn_readlane(lanesum, 16);

    __shared__ int red[8];
    if ((tid & 31) == 0) red[tid >> 5] = wavesum;
    __syncthreads();

    int num_trues = 0;
#pragma unroll
    for (int w = 0; w < 8; ++w) num_trues += red[w];
    num_trues = __builtin_amdgcn_readfirstlane(num_trues);

    // Uniform window computation (floor-division semantics; v >= -1 always).
    const int page = page_size_ptr[0];
    const int seq = seq_lens[row];
    const int v = seq + (D - num_trues) - 1;
    int q;
    if ((page & (page - 1)) == 0) {
        // page is a power of two (the real case: 64): (v+page)>>log2 is exact
        // floor since v+page >= page-1 >= 0.
        const int lg = 31 - __clz(page);
        q = ((v + page) >> lg) - 1;
    } else {
        q = (v + page) / page - 1;             // floor(v/page) for v >= -page
    }
    const int start = q * page - seq;
    const int lo = start > 0 ? start : 0;
    int hi = start + page;
    if (hi > D) hi = D;

#pragma unroll
    for (int c = 0; c < 8; ++c) {
        const int d0 = c * 1024 + tid * 4;
        const unsigned p = pk[c];
        v4f f;
        f.x = (((p      ) & 1u) && !((d0    ) >= lo && (d0    ) < hi)) ? 1.0f : 0.0f;
        f.y = (((p >>  8) & 1u) && !((d0 + 1) >= lo && (d0 + 1) < hi)) ? 1.0f : 0.0f;
        f.z = (((p >> 16) & 1u) && !((d0 + 2) >= lo && (d0 + 2) < hi)) ? 1.0f : 0.0f;
        f.w = (((p >> 24) & 1u) && !((d0 + 3) >= lo && (d0 + 3) < hi)) ? 1.0f : 0.0f;
        __builtin_nontemporal_store(f, &orow[c * 256 + tid]);
    }
}

extern "C" void kernel_launch(void* const* d_in, const int* in_sizes, int n_in,
                              void* d_out, int out_size, void* d_ws, size_t ws_size,
                              hipStream_t stream) {
    const int* seq_lens  = (const int*)d_in[0];   // [B] int32
    const int* mask      = (const int*)d_in[1];   // [B, D] bool as int32
    const int* page_size = (const int*)d_in[2];   // scalar int
    float* out = (float*)d_out;                   // [B, D] float 0/1

    const int B = in_sizes[0];
    const int D = in_sizes[1] / B;                // 8192 (kernel layout assumes this)

    evict_mask_page_clear<<<B, 256, 0, stream>>>(seq_lens, mask, page_size, out, D);
}
